// BiLSTM_CRF_77146202571330
// MI455X (gfx1250) — compile-verified, matrix operand order checked by reading, not run on hardware
//
#include <hip/hip_runtime.h>

// ---------------------------------------------------------------------------
// BiLSTM-CRF forward for MI455X (gfx1250, wave32).
//   - bf16 WMMA (v_wmma_f32_16x16x32_bf16) for all GEMMs, fp32 accumulation
//   - persistent recurrent kernel with per-timestep device-scope grid sync
//   - TDM (tensor_load_to_lds) stages h_prev into LDS each timestep
//   - fused encoder+emit kernel staging tanh states in LDS
//   - fp32 scalar CRF, one block per batch element
// ---------------------------------------------------------------------------

#define T_DIM 512
#define B_DIM 64
#define E_DIM 256
#define H_DIM 512
#define G_DIM 2048          // 4*H
#define KCAT  768           // E + H
#define KTAG  50
#define KPAD  64
#define NROW  (T_DIM * B_DIM)   // 32768
#define NCHUNK 16           // hidden-dim blocks per direction in LSTM kernel

typedef __attribute__((ext_vector_type(8)))  float        v8f;
typedef __attribute__((ext_vector_type(16))) __bf16       v16bf;
typedef __attribute__((ext_vector_type(4)))  unsigned int v4u;
typedef __attribute__((ext_vector_type(8)))  unsigned int v8u;
typedef __attribute__((ext_vector_type(4)))  unsigned int u32x4;
typedef __attribute__((ext_vector_type(8)))  int          i32x8;
typedef __attribute__((ext_vector_type(4)))  int          i32x4;

__device__ __forceinline__ unsigned short f2bf(float f) {
  unsigned u = __float_as_uint(f);
  unsigned r = ((u >> 16) & 1u) + 0x7FFFu;      // round-to-nearest-even
  return (unsigned short)((u + r) >> 16);
}
__device__ __forceinline__ float sigmoidf_(float x) { return 1.0f / (1.0f + expf(-x)); }

// Load a 16(row) x 32(K) bf16 tile (row-major, K contiguous) into the wave32
// WMMA A/B operand layout:
//   lane L<16 : row L,    K = {0..7} and {16..23}
//   lane L>=16: row L-16, K = {8..15} and {24..31}
// Two aligned 16B loads per lane (strides are multiples of 8 elements).
// NOTE: keep callers' pointer provenance unambiguous (pure global OR pure LDS)
// so address-space inference emits global_load_b128 / ds_load_b128, not flat.
__device__ __forceinline__ v16bf load_tile(const unsigned short* base, int stride, int lane) {
  const int r = lane & 15, hi = lane >> 4;
  const unsigned short* p = base + (size_t)r * stride + hi * 8;
  v4u lo = *reinterpret_cast<const v4u*>(p);
  v4u hh = *reinterpret_cast<const v4u*>(p + 16);
  v8u u;
  u[0] = lo[0]; u[1] = lo[1]; u[2] = lo[2]; u[3] = lo[3];
  u[4] = hh[0]; u[5] = hh[1]; u[6] = hh[2]; u[7] = hh[3];
  return __builtin_bit_cast(v16bf, u);
}

__device__ __forceinline__ v8f wmma_bf16(v16bf a, v16bf b, v8f c) {
  return __builtin_amdgcn_wmma_f32_16x16x32_bf16(false, a, false, b, (short)0, c, false, false);
}
__device__ __forceinline__ v8f zero8() { v8f z = {0.f,0.f,0.f,0.f,0.f,0.f,0.f,0.f}; return z; }

// ------------------------------- prep kernels ------------------------------

__global__ void reset_kernel(unsigned int* syncp, unsigned short* hbuf) {
  int idx = blockIdx.x * blockDim.x + threadIdx.x;
  if (idx < 8) syncp[idx] = 0u;
  for (int i = idx; i < 2 * 2 * B_DIM * H_DIM; i += gridDim.x * blockDim.x) hbuf[i] = 0;
}

// wcat[n][k] = k<E ? w_ih[n][k] : w_hh[n][k-E], bf16
__global__ void cat_weights_kernel(const float* __restrict__ wih,
                                   const float* __restrict__ whh,
                                   unsigned short* __restrict__ wcat) {
  for (int idx = blockIdx.x * blockDim.x + threadIdx.x; idx < G_DIM * KCAT;
       idx += gridDim.x * blockDim.x) {
    int n = idx / KCAT, k = idx % KCAT;
    float v = (k < E_DIM) ? wih[(size_t)n * E_DIM + k] : whh[(size_t)n * H_DIM + (k - E_DIM)];
    wcat[idx] = f2bf(v);
  }
}

__global__ void conv_bf16_kernel(const float* __restrict__ src, unsigned short* __restrict__ dst, int n) {
  for (int i = blockIdx.x * blockDim.x + threadIdx.x; i < n; i += gridDim.x * blockDim.x)
    dst[i] = f2bf(src[i]);
}

// pad w_out [50,512] -> [64,512] bf16 (rows >=50 are zero)
__global__ void pad_wout_kernel(const float* __restrict__ wout, unsigned short* __restrict__ dst) {
  for (int idx = blockIdx.x * blockDim.x + threadIdx.x; idx < KPAD * H_DIM;
       idx += gridDim.x * blockDim.x) {
    int n = idx / H_DIM, k = idx % H_DIM;
    dst[idx] = f2bf(n < KTAG ? wout[(size_t)n * H_DIM + k] : 0.f);
  }
}

// xe[(t*B+b)][e] = bf16(embed[tokens[t*B+b]][e])
__global__ void embed_kernel(const int* __restrict__ tokens, const float* __restrict__ embed,
                             unsigned short* __restrict__ xe) {
  for (int idx = blockIdx.x * blockDim.x + threadIdx.x; idx < NROW * E_DIM;
       idx += gridDim.x * blockDim.x) {
    int r = idx >> 8, e = idx & (E_DIM - 1);
    int tok = tokens[r];
    xe[idx] = f2bf(embed[(size_t)tok * E_DIM + e]);
  }
}

// ------------------------- persistent BiLSTM kernel ------------------------
// grid = (16, 2): x = hidden chunk (32 units), y = direction. block = 256 (8 waves).
// Per step: wave 0 issues one TDM tensor_load_to_lds to DMA h_prev (64x512 bf16,
// 64 KB) into LDS, then each wave computes one 16-col gate N-tile over 4 M-tiles
// (B=64): x-part (K=256) A from global, h-part (K=512) A from LDS via ds_load.
// Gates are exchanged through LDS (aliasing the h staging buffer — disjoint
// phases, barrier-separated); h exchanged through a double-buffered global
// array with an agent-scope arrive/spin grid sync.

__global__ __launch_bounds__(256)
void lstm_kernel(const unsigned short* __restrict__ wcat_f,
                 const unsigned short* __restrict__ wcat_b,
                 const float* __restrict__ b_f, const float* __restrict__ b_b,
                 const unsigned short* __restrict__ xe,
                 unsigned short* __restrict__ hf, unsigned short* __restrict__ hb,
                 unsigned short* __restrict__ hbuf,     // [dir][2][B][H]
                 unsigned int* __restrict__ syncp) {    // [dir]
  const int dir = blockIdx.y;
  const int cb  = blockIdx.x;                // hidden chunk 0..15
  const int tid = threadIdx.x;
  const int lane = tid & 31;
  const int w    = tid >> 5;                 // wave 0..7
  const int gt   = w >> 1;                   // gate type: 0=i 1=f 2=g 3=o
  const int jsub = (w & 1) * 16;

  const unsigned short* wcat = dir ? wcat_b : wcat_f;
  const float* bias          = dir ? b_b : b_f;
  unsigned short* hs         = dir ? hb : hf;
  unsigned int* cnt          = syncp + dir;
  unsigned short* hb0        = hbuf + (size_t)dir * 2 * B_DIM * H_DIM;

  const int nrow0   = gt * H_DIM + cb * 32 + jsub;   // first weight row of this wave's tile
  const float biasv = bias[nrow0 + (lane & 15)];
  const unsigned short* Wb = wcat + (size_t)nrow0 * KCAT;

  // 64 KB LDS, time-shared within each step:
  //   phase A (TDM + GEMM): bf16 h_prev staging [64][512]
  //   phase B (cell update): fp32 gate exchange [64][128] (first 32 KB)
  __shared__ __align__(16) unsigned int lds_raw[16384];
  unsigned short* hstage = reinterpret_cast<unsigned short*>(lds_raw);
  float*          gates  = reinterpret_cast<float*>(lds_raw);
  const unsigned lds_base = (unsigned)(size_t)(void*)lds_raw;  // LDS byte offset

  // Per-thread cell state: fixed jloc = tid&31, 8 batch rows each.
  float creg[8];
#pragma unroll
  for (int r = 0; r < 8; ++r) creg[r] = 0.f;
  const int jloc  = tid & 31;
  const int bbase = (tid >> 5) * 8;
  const int jglob = cb * 32 + jloc;

  for (int step = 0; step < T_DIM; ++step) {
    const int tproc = dir ? (T_DIM - 1 - step) : step;
    const int rp = step & 1;
    const unsigned short* hprev = hb0 + (size_t)rp * (B_DIM * H_DIM);
    unsigned short* hnext       = hb0 + (size_t)(1 - rp) * (B_DIM * H_DIM);

    // --- TDM: DMA h_prev (64 rows x 512 cols, 2B elements) global -> LDS ---
    if (w == 0) {
      unsigned long long ga = (unsigned long long)(size_t)(const void*)hprev;
      u32x4 g0;
      g0[0] = 1u;                                            // count=1 (valid user D#)
      g0[1] = lds_base;                                      // lds_addr
      g0[2] = (unsigned)ga;                                  // global_addr[31:0]
      g0[3] = (unsigned)((ga >> 32) & 0x01FFFFFFu) | 0x80000000u; // addr[56:32] | type=2
      i32x8 g1;
      g1[0] = 0x00010000;            // data_size=1 (2 bytes); no multicast/pad/iterate
      g1[1] = (int)(512u << 16);     // tensor_dim0[15:0]=512 at bits[63:48]
      g1[2] = (int)(64u << 16);      // tensor_dim0[31:16]=0, tensor_dim1[15:0]=64
      g1[3] = (int)(512u << 16);     // tensor_dim1[31:16]=0, tile_dim0=512
      g1[4] = 64;                    // tile_dim1=64, tile_dim2=0 (2-D tile)
      g1[5] = 512;                   // tensor_dim0_stride[31:0]=512
      g1[6] = 0;                     // stride0[47:32]=0, stride1[15:0]=0
      g1[7] = 0;
      i32x4 gz4 = {0, 0, 0, 0};                  // groups 2/3 unused for 2-D tiles
      i32x8 gz8 = {0, 0, 0, 0, 0, 0, 0, 0};      // extra operand (6-arg toolchain form)
      __builtin_amdgcn_tensor_load_to_lds(g0, g1, gz4, gz4, gz8, 0);
      __builtin_amdgcn_s_wait_tensorcnt((short)0);
    }
    __syncthreads();

    v8f acc[4];
#pragma unroll
    for (int mt = 0; mt < 4; ++mt) acc[mt] = zero8();

    const unsigned short* xrow = xe + (size_t)tproc * B_DIM * E_DIM;

    // x-part: K = 0..255, A tiles from global (pure global provenance).
#pragma unroll 4
    for (int kc = 0; kc < E_DIM / 32; ++kc) {
      const int kb = kc * 32;
      v16bf bmat = load_tile(Wb + kb, KCAT, lane);
#pragma unroll
      for (int mt = 0; mt < 4; ++mt) {
        v16bf amat = load_tile(xrow + (size_t)mt * 16 * E_DIM + kb, E_DIM, lane);
        acc[mt] = wmma_bf16(amat, bmat, acc[mt]);
      }
    }

    // h-part: K = 256..767, A tiles from LDS (pure LDS provenance -> ds_load_b128).
#pragma unroll 4
    for (int kc = 0; kc < H_DIM / 32; ++kc) {
      const int kb = kc * 32;
      v16bf bmat = load_tile(Wb + E_DIM + kb, KCAT, lane);
#pragma unroll
      for (int mt = 0; mt < 4; ++mt) {
        v16bf amat = load_tile(hstage + mt * 16 * H_DIM + kb, H_DIM, lane);
        acc[mt] = wmma_bf16(amat, bmat, acc[mt]);
      }
    }
    __syncthreads();   // all waves done reading hstage before gates alias it

    // C/D layout: vgpr i, lanes 0-15 -> row mt*16+i, lanes 16-31 -> row mt*16+8+i; col = lane&15.
    const int col = w * 16 + (lane & 15);    // = gt*32 + jsub + (lane&15)
    const int rhi = (lane >> 4) * 8;
#pragma unroll
    for (int mt = 0; mt < 4; ++mt)
#pragma unroll
      for (int i = 0; i < 8; ++i)
        gates[(mt * 16 + rhi + i) * 128 + col] = acc[mt][i] + biasv;
    __syncthreads();

    // Elementwise LSTM cell (torch gate order i,f,g,o).
#pragma unroll
    for (int r = 0; r < 8; ++r) {
      const int b = bbase + r;
      float iv = sigmoidf_(gates[b * 128 +  0 + jloc]);
      float fv = sigmoidf_(gates[b * 128 + 32 + jloc]);
      float gv = tanhf(    gates[b * 128 + 64 + jloc]);
      float ov = sigmoidf_(gates[b * 128 + 96 + jloc]);
      float c = fv * creg[r] + iv * gv;
      creg[r] = c;
      float h = ov * tanhf(c);
      unsigned short hq = f2bf(h);
      hnext[b * H_DIM + jglob] = hq;
      hs[((size_t)tproc * B_DIM + b) * H_DIM + jglob] = hq;
    }

    // Grid-wide sync across the 16 blocks of this direction.
    __threadfence();
    __syncthreads();
    if (tid == 0) {
      __hip_atomic_fetch_add(cnt, 1u, __ATOMIC_RELEASE, __HIP_MEMORY_SCOPE_AGENT);
      const unsigned target = (unsigned)NCHUNK * (unsigned)(step + 1);
      while (__hip_atomic_load(cnt, __ATOMIC_ACQUIRE, __HIP_MEMORY_SCOPE_AGENT) < target)
        __builtin_amdgcn_s_sleep(1);
    }
    __syncthreads();
  }
}

// ---------------------- fused encoder + emit kernel ------------------------
// One block (512 threads = 16 waves) per 64 rows of [T*B].
// GEMM1: states = tanh([hf|hb] @ w_enc^T + b_enc)  (M=64,N=512,K=1024) -> LDS bf16
// GEMM2: emit   = states @ w_out^T + b_out         (M=64,N=64pad,K=512)

__global__ __launch_bounds__(512)
void enc_emit_kernel(const unsigned short* __restrict__ hf,
                     const unsigned short* __restrict__ hb,
                     const unsigned short* __restrict__ wenc,
                     const float* __restrict__ benc,
                     const unsigned short* __restrict__ wout,
                     const float* __restrict__ bout,
                     float* __restrict__ emit) {
  const int tid = threadIdx.x, lane = tid & 31, w = tid >> 5;   // 16 waves
  const size_t rowbase = (size_t)blockIdx.x * 64;
  __shared__ __align__(16) unsigned short states[64 * H_DIM];   // 64 KB

  for (int nt2 = 0; nt2 < 2; ++nt2) {
    const int ntile = w * 2 + nt2;           // 0..31
    const int ncol0 = ntile * 16;
    const float bv = benc[ncol0 + (lane & 15)];
    v8f acc[4];
#pragma unroll
    for (int mt = 0; mt < 4; ++mt) acc[mt] = zero8();
    // hf half (K = 0..511) — separate loop keeps pointer provenance unmixed.
#pragma unroll 4
    for (int kc = 0; kc < 16; ++kc) {
      const int kb = kc * 32;
      v16bf bmat = load_tile(wenc + (size_t)ncol0 * (2 * H_DIM) + kb, 2 * H_DIM, lane);
#pragma unroll
      for (int mt = 0; mt < 4; ++mt) {
        v16bf amat = load_tile(hf + (rowbase + mt * 16) * H_DIM + kb, H_DIM, lane);
        acc[mt] = wmma_bf16(amat, bmat, acc[mt]);
      }
    }
    // hb half (K = 512..1023)
#pragma unroll 4
    for (int kc = 0; kc < 16; ++kc) {
      const int kb = kc * 32;
      v16bf bmat = load_tile(wenc + (size_t)ncol0 * (2 * H_DIM) + H_DIM + kb, 2 * H_DIM, lane);
#pragma unroll
      for (int mt = 0; mt < 4; ++mt) {
        v16bf amat = load_tile(hb + (rowbase + mt * 16) * H_DIM + kb, H_DIM, lane);
        acc[mt] = wmma_bf16(amat, bmat, acc[mt]);
      }
    }
    const int rhi = (lane >> 4) * 8, c = lane & 15;
#pragma unroll
    for (int mt = 0; mt < 4; ++mt)
#pragma unroll
      for (int i = 0; i < 8; ++i)
        states[(mt * 16 + rhi + i) * H_DIM + ncol0 + c] = f2bf(tanhf(acc[mt][i] + bv));
  }
  __syncthreads();

  // GEMM2: 16 waves <-> 4x4 output tiles. A from LDS (ds_load), B from global.
  const int mt = w >> 2, nt = w & 3;
  v8f acc = zero8();
#pragma unroll 4
  for (int kc = 0; kc < 16; ++kc) {
    v16bf amat = load_tile(&states[(mt * 16) * H_DIM + kc * 32], H_DIM, lane);
    v16bf bmat = load_tile(wout + (size_t)(nt * 16) * H_DIM + kc * 32, H_DIM, lane);
    acc = wmma_bf16(amat, bmat, acc);
  }
  const int col = nt * 16 + (lane & 15);
  const float bv = (col < KTAG) ? bout[col] : 0.f;
  const int rhi = (lane >> 4) * 8;
#pragma unroll
  for (int i = 0; i < 8; ++i)
    emit[(rowbase + mt * 16 + rhi + i) * KPAD + col] = acc[i] + bv;
}

// ------------------------------- CRF kernel --------------------------------
// One block per batch element; thread j computes alpha'[j] (fp32 throughout).

__global__ __launch_bounds__(64)
void crf_kernel(const float* __restrict__ emit, const float* __restrict__ trans,
                float* __restrict__ out) {
  const int b = blockIdx.x;
  const int j = threadIdx.x;
  __shared__ float transL[KTAG * KTAG];
  __shared__ float alpha[KPAD];
  __shared__ float nalpha[KPAD];

  for (int idx = j; idx < KTAG * KTAG; idx += 64) transL[idx] = trans[idx];
  if (j < KTAG) alpha[j] = (j == 0) ? 0.f : -10000.f;   // START = 0
  __syncthreads();

  for (int t = 0; t < T_DIM; ++t) {
    if (j < KTAG) {
      const float* tr = &transL[j * KTAG];
      float m = -3.4e38f;
#pragma unroll 5
      for (int i = 0; i < KTAG; ++i) m = fmaxf(m, alpha[i] + tr[i]);
      float s = 0.f;
#pragma unroll 5
      for (int i = 0; i < KTAG; ++i) s += expf(alpha[i] + tr[i] - m);
      nalpha[j] = m + logf(s) + emit[((size_t)t * B_DIM + b) * KPAD + j];
    }
    __syncthreads();
    if (j < KTAG) alpha[j] = nalpha[j];
    __syncthreads();
  }

  if (j == 0) {                                         // STOP = KTAG-1
    const float* tr = &transL[(KTAG - 1) * KTAG];
    float m = -3.4e38f;
    for (int i = 0; i < KTAG; ++i) m = fmaxf(m, alpha[i] + tr[i]);
    float s = 0.f;
    for (int i = 0; i < KTAG; ++i) s += expf(alpha[i] + tr[i] - m);
    out[b] = m + logf(s);
  }
}

// ------------------------------- launcher ----------------------------------

extern "C" void kernel_launch(void* const* d_in, const int* in_sizes, int n_in,
                              void* d_out, int out_size, void* d_ws, size_t ws_size,
                              hipStream_t stream) {
  const int*   tokens = (const int*)  d_in[0];
  const float* embed  = (const float*)d_in[1];
  const float* w_ih_f = (const float*)d_in[2];
  const float* w_hh_f = (const float*)d_in[3];
  const float* b_f    = (const float*)d_in[4];
  const float* w_ih_b = (const float*)d_in[5];
  const float* w_hh_b = (const float*)d_in[6];
  const float* b_b    = (const float*)d_in[7];
  const float* w_enc  = (const float*)d_in[8];
  const float* b_enc  = (const float*)d_in[9];
  const float* w_out  = (const float*)d_in[10];
  const float* b_out  = (const float*)d_in[11];
  const float* trans  = (const float*)d_in[12];
  float* out = (float*)d_out;

  char* ws = (char*)d_ws;
  size_t off = 0;
  auto take = [&](size_t bytes) -> char* {
    char* p = ws + off;
    off += (bytes + 255) & ~(size_t)255;
    return p;
  };
  unsigned int*   syncp  = (unsigned int*)  take(256);
  unsigned short* wcat_f = (unsigned short*)take((size_t)G_DIM * KCAT * 2);      // 3 MB
  unsigned short* wcat_b = (unsigned short*)take((size_t)G_DIM * KCAT * 2);      // 3 MB
  unsigned short* wencb  = (unsigned short*)take((size_t)H_DIM * 2 * H_DIM * 2); // 1 MB
  unsigned short* woutb  = (unsigned short*)take((size_t)KPAD * H_DIM * 2);      // 64 KB
  unsigned short* xe     = (unsigned short*)take((size_t)NROW * E_DIM * 2);      // 16 MB
  unsigned short* hf     = (unsigned short*)take((size_t)NROW * H_DIM * 2);      // 32 MB
  unsigned short* hb     = (unsigned short*)take((size_t)NROW * H_DIM * 2);      // 32 MB
  unsigned short* hbuf   = (unsigned short*)take((size_t)2 * 2 * B_DIM * H_DIM * 2);
  float*          emit   = (float*)         take((size_t)NROW * KPAD * 4);       // 8 MB
  (void)ws_size; (void)in_sizes; (void)n_in; (void)out_size;                     // ~96 MB total

  reset_kernel<<<dim3(256), dim3(256), 0, stream>>>(syncp, hbuf);
  cat_weights_kernel<<<dim3(1024), dim3(256), 0, stream>>>(w_ih_f, w_hh_f, wcat_f);
  cat_weights_kernel<<<dim3(1024), dim3(256), 0, stream>>>(w_ih_b, w_hh_b, wcat_b);
  conv_bf16_kernel<<<dim3(512), dim3(256), 0, stream>>>(w_enc, wencb, H_DIM * 2 * H_DIM);
  pad_wout_kernel<<<dim3(64), dim3(256), 0, stream>>>(w_out, woutb);
  embed_kernel<<<dim3(4096), dim3(256), 0, stream>>>(tokens, embed, xe);

  lstm_kernel<<<dim3(NCHUNK, 2), dim3(256), 0, stream>>>(
      wcat_f, wcat_b, b_f, b_b, xe, hf, hb, hbuf, syncp);

  enc_emit_kernel<<<dim3(NROW / 64), dim3(512), 0, stream>>>(
      hf, hb, wencb, b_enc, woutb, b_out, emit);

  crf_kernel<<<dim3(B_DIM), dim3(64), 0, stream>>>(emit, trans, out);
}
